// MaskExtractor_47038481826292
// MI455X (gfx1250) — matrix operand order repaired and checked by reading, not compiled, not run on hardware
//
#include <hip/hip_runtime.h>
#include <stdint.h>
#include <stddef.h>

// ---------------------------------------------------------------------------
// MaskExtractor for MI455X (gfx1250, wave32, WMMA + TDM).
// Pipeline: pos-add -> threefry/top-k kmeans init -> 3x kmeans (TDM-staged
// centroids) -> 2x (LN + low-rank MHA via bf16 WMMA) -> GELU FFN (bf16 WMMA).
// ---------------------------------------------------------------------------

typedef __attribute__((ext_vector_type(16))) __bf16 v16bf;
typedef __attribute__((ext_vector_type(8)))  float  v8f;

#define R_   16
#define NT_  2304     // 48*48 tokens per region
#define C_   1024
#define H_   8
#define DH_  128
#define K_   16       // mask_num / clusters
#define HID_ 4096

// ------------------------------- utilities ---------------------------------

__device__ inline uint32_t pack_bf16(float lo, float hi) {
  union { __bf16 b[2]; uint32_t u; } u;
  u.b[0] = (__bf16)lo; u.b[1] = (__bf16)hi;
  return u.u;
}

__global__ void f32_to_bf16_kernel(const float* __restrict__ in,
                                   __bf16* __restrict__ out, long long n) {
  long long i = (long long)blockIdx.x * blockDim.x + threadIdx.x;
  if (i < n) out[i] = (__bf16)in[i];
}

// f = feats + pos @ pos_w + pos_b
__global__ void pos_tokens_kernel(const float* __restrict__ feats,
                                  const float* __restrict__ boxes,
                                  const float* __restrict__ pos_w,
                                  const float* __restrict__ pos_b,
                                  float* __restrict__ tokens, long long total) {
  long long idx = (long long)blockIdx.x * blockDim.x + threadIdx.x;
  if (idx >= total) return;
  int c = (int)(idx & (C_ - 1));
  long long t = idx >> 10;              // r*2304 + y*48 + x
  int x = (int)(t % 48);
  int y = (int)((t / 48) % 48);
  int r = (int)(t / NT_);
  const float* box = boxes + r * 4;
  float xg = ((x * (1.0f / 48.0f)) * (box[2] - box[0]) + box[0]) * (1.0f / 999.0f);
  float yg = ((y * (1.0f / 48.0f)) * (box[3] - box[1]) + box[1]) * (1.0f / 999.0f);
  tokens[idx] = feats[idx] + xg * pos_w[c] + yg * pos_w[C_ + c] + pos_b[c];
}

// ------------------ Tensor Data Mover: 2-D tile -> LDS ----------------------
// Hand-packed D# per CDNA5 ISA ch.8 (group0: count/lds/global/type,
// group1: data_size, tensor dims, tile dims, dim0 stride).
// This toolchain exposes the 6-arg builtin:
//   (uint32x4 g0, int32x8 g1, int32x4 g2, int32x4 g3, int32x8 gx, i32 cpol)

typedef uint32_t u32x4_t __attribute__((ext_vector_type(4)));
typedef int      i32x8_t __attribute__((ext_vector_type(8)));
typedef int      i32x4_t __attribute__((ext_vector_type(4)));

__device__ inline void tdm_load_2d(uint32_t lds_byte_off, const void* gptr,
                                   uint32_t td0, uint32_t td1,
                                   uint32_t tile0, uint32_t tile1,
                                   uint64_t stride0, uint32_t dsize_code) {
  uint64_t ga = (uint64_t)gptr;
  u32x4_t g0;
  g0[0] = 1u;                                        // count=1 (valid user D#)
  g0[1] = lds_byte_off;                              // lds_addr
  g0[2] = (uint32_t)ga;                              // global_addr[31:0]
  g0[3] = ((uint32_t)(ga >> 32) & 0x01FFFFFFu)       // global_addr[56:32]
          | (2u << 30);                              // type=2 ("image")
  i32x8_t g1;
  g1[0] = (int)(dsize_code << 16);                   // wg_mask=0, data_size
  g1[1] = (int)((td0 & 0xFFFFu) << 16);              // tensor_dim0[15:0]
  g1[2] = (int)((td0 >> 16) | ((td1 & 0xFFFFu) << 16));  // td0 hi | td1 lo
  g1[3] = (int)((td1 >> 16) | (tile0 << 16));        // td1 hi | tile_dim0
  g1[4] = (int)tile1;                                // tile_dim1 (tile_dim2=0)
  g1[5] = (int)(uint32_t)stride0;                    // tensor_dim0_stride lo
  g1[6] = (int)(uint32_t)(stride0 >> 32);            // stride hi | td1_stride=0
  g1[7] = 0;
  i32x4_t z4 = {0, 0, 0, 0};                         // groups 2/3 unused (2-D)
  i32x8_t z8 = {0, 0, 0, 0, 0, 0, 0, 0};
  __builtin_amdgcn_tensor_load_to_lds(g0, g1, z4, z4, z8, 0);
}

// ------------------------- threefry2x32 (JAX PRNG) -------------------------

__device__ inline void threefry2x32(uint32_t k0, uint32_t k1,
                                    uint32_t x0, uint32_t x1,
                                    uint32_t* o0, uint32_t* o1) {
  uint32_t ks[3] = {k0, k1, k0 ^ k1 ^ 0x1BD11BDAu};
  const int r0[4] = {13, 15, 26, 6};
  const int r1[4] = {17, 29, 16, 24};
  x0 += ks[0]; x1 += ks[1];
  #pragma unroll
  for (int i = 0; i < 5; ++i) {
    const int* rot = (i & 1) ? r1 : r0;
    #pragma unroll
    for (int j = 0; j < 4; ++j) {
      x0 += x1;
      x1 = (x1 << rot[j]) | (x1 >> (32 - rot[j]));
      x1 ^= x0;
    }
    x0 += ks[(i + 1) % 3];
    x1 += ks[(i + 2) % 3] + (uint32_t)(i + 1);
  }
  *o0 = x0; *o1 = x1;
}

// u[r,t] = valid ? uniform(key_r)[t] : -1   (key_r = split(key(42), 16)[r])
__global__ __launch_bounds__(256)
void uniform_kernel(const int* __restrict__ masks, float* __restrict__ u) {
  int r = blockIdx.x;
  int tid = threadIdx.x;
  uint32_t k0, k1, a, b;
  uint32_t j0 = 2u * r, j1 = 2u * r + 1u;
  if (r < 8) {   // split: counts=iota(32), halves (i, 16+i); out[2r],out[2r+1]
    threefry2x32(0u, 42u, j0, 16u + j0, &a, &b); k0 = a;
    threefry2x32(0u, 42u, j1, 16u + j1, &a, &b); k1 = a;
  } else {
    threefry2x32(0u, 42u, j0 - 16u, j0, &a, &b); k0 = b;
    threefry2x32(0u, 42u, j1 - 16u, j1, &a, &b); k1 = b;
  }
  for (int t = tid; t < NT_; t += 256) {
    uint32_t o0, o1, bits;
    if (t < NT_ / 2) { threefry2x32(k0, k1, (uint32_t)t, (uint32_t)(NT_ / 2 + t), &o0, &o1); bits = o0; }
    else             { threefry2x32(k0, k1, (uint32_t)(t - NT_ / 2), (uint32_t)t, &o0, &o1); bits = o1; }
    float f = __uint_as_float((bits >> 9) | 0x3f800000u) - 1.0f;
    u[r * NT_ + t] = (masks[r * NT_ + t] > 0) ? f : -1.0f;
  }
}

// Iterative top-16 argmax (matches lax.top_k ordering), gather init centroids.
__global__ __launch_bounds__(256)
void topk_init_kernel(float* __restrict__ u, const float* __restrict__ tokens,
                      float* __restrict__ cent) {
  int r = blockIdx.x, tid = threadIdx.x;
  __shared__ float sv[256];
  __shared__ int   si[256];
  __shared__ int   chosen[K_];
  float* ur = u + r * NT_;
  for (int p = 0; p < K_; ++p) {
    float bv = -1e30f; int bi = 0;
    for (int t = tid; t < NT_; t += 256) {
      float v = ur[t];
      if (v > bv) { bv = v; bi = t; }
    }
    sv[tid] = bv; si[tid] = bi;
    __syncthreads();
    for (int s = 128; s > 0; s >>= 1) {
      if (tid < s) {
        if (sv[tid + s] > sv[tid] ||
            (sv[tid + s] == sv[tid] && si[tid + s] < si[tid])) {
          sv[tid] = sv[tid + s]; si[tid] = si[tid + s];
        }
      }
      __syncthreads();
    }
    if (tid == 0) { chosen[p] = si[0]; ur[si[0]] = -3.0f; }
    __syncthreads();
  }
  for (int i = tid; i < K_ * C_; i += 256) {
    int p = i >> 10, c = i & (C_ - 1);
    cent[((long long)r * K_ + p) * C_ + c] =
        tokens[((long long)r * NT_ + chosen[p]) * C_ + c];
  }
}

// ------------------------------- k-means -----------------------------------

// labels = argmin_k ( |c_k|^2 - 2 t.c_k ); centroid tile staged via TDM.
__global__ __launch_bounds__(256)
void kmeans_assign_kernel(const float* __restrict__ tokens,
                          const float* __restrict__ cent,
                          int* __restrict__ labels) {
  __shared__ __align__(16) float cl[K_ * 512];
  __shared__ float csq[K_];
  int r = blockIdx.x, blk = blockIdx.y, tid = threadIdx.x;
  const float* cr = cent + (long long)r * K_ * C_;
  int t = blk * 256 + tid;
  const float* tok = tokens + ((long long)r * NT_ + t) * C_;
  float acc[K_];
  #pragma unroll
  for (int k = 0; k < K_; ++k) acc[k] = 0.0f;
  if (tid < K_) csq[tid] = 0.0f;
  __syncthreads();
  for (int ch = 0; ch < 2; ++ch) {
    if (tid == 0) {
      // DMA 16x512 f32 tile (row stride C_) into LDS, compacted.
      tdm_load_2d((uint32_t)(size_t)cl, cr + ch * 512,
                  /*tensor*/ C_, K_, /*tile*/ 512, K_,
                  /*stride0*/ C_, /*data_size code: 4B*/ 2u);
      __builtin_amdgcn_s_wait_tensorcnt(0);
    }
    __syncthreads();
    if (tid < K_) {
      float s = 0.0f;
      for (int c = 0; c < 512; ++c) { float v = cl[tid * 512 + c]; s += v * v; }
      csq[tid] += s;
    }
    for (int c = 0; c < 512; ++c) {
      float tv = tok[ch * 512 + c];
      #pragma unroll
      for (int k = 0; k < K_; ++k) acc[k] += tv * cl[k * 512 + c];
    }
    __syncthreads();
  }
  int best = 0; float bd = csq[0] - 2.0f * acc[0];
  #pragma unroll
  for (int k = 1; k < K_; ++k) {
    float d = csq[k] - 2.0f * acc[k];
    if (d < bd) { bd = d; best = k; }
  }
  labels[r * NT_ + t] = best;
}

// segment sums of valid tokens into sums/cnts (grid: R x 8 chunks x 4 slabs)
__global__ __launch_bounds__(128)
void kmeans_update_kernel(const float* __restrict__ tokens,
                          const int* __restrict__ labels,
                          const int* __restrict__ masks,
                          float* __restrict__ sums, float* __restrict__ cnts) {
  int r = blockIdx.x, ch = blockIdx.y, sl = blockIdx.z, tid = threadIdx.x;
  __shared__ float acc[K_ * 128];
  __shared__ float lcnt[K_];
  for (int i = tid; i < K_ * 128; i += 128) acc[i] = 0.0f;
  if (tid < K_) lcnt[tid] = 0.0f;
  __syncthreads();
  int t0 = sl * (NT_ / 4);
  for (int t = t0; t < t0 + NT_ / 4; ++t) {
    int lab = labels[r * NT_ + t];
    if (masks[r * NT_ + t] > 0) {
      acc[lab * 128 + tid] += tokens[((long long)r * NT_ + t) * C_ + ch * 128 + tid];
      if (tid == 0) lcnt[lab] += 1.0f;
    }
  }
  __syncthreads();
  for (int k = 0; k < K_; ++k)
    atomicAdd(&sums[((long long)r * K_ + k) * C_ + ch * 128 + tid], acc[k * 128 + tid]);
  if (ch == 0 && tid < K_) atomicAdd(&cnts[r * K_ + tid], lcnt[tid]);
}

__global__ void cent_update_kernel(const float* __restrict__ sums,
                                   const float* __restrict__ cnts,
                                   float* __restrict__ cent, int total) {
  int idx = blockIdx.x * blockDim.x + threadIdx.x;
  if (idx >= total) return;
  float c = cnts[idx >> 10];
  if (c > 0.0f) cent[idx] = sums[idx] / fmaxf(c, 1.0f);
}

// ---------------------------- LN / softmax ---------------------------------

__global__ __launch_bounds__(256)
void layernorm_kernel(const float* __restrict__ in, const float* __restrict__ g,
                      const float* __restrict__ b, float* __restrict__ out) {
  int row = blockIdx.x, tid = threadIdx.x;
  __shared__ float red[256];
  const float* x = in + (long long)row * C_;
  float s = 0.0f;
  for (int c = tid; c < C_; c += 256) s += x[c];
  red[tid] = s; __syncthreads();
  for (int st = 128; st > 0; st >>= 1) { if (tid < st) red[tid] += red[tid + st]; __syncthreads(); }
  float mean = red[0] * (1.0f / C_);
  __syncthreads();
  float s2 = 0.0f;
  for (int c = tid; c < C_; c += 256) { float d = x[c] - mean; s2 += d * d; }
  red[tid] = s2; __syncthreads();
  for (int st = 128; st > 0; st >>= 1) { if (tid < st) red[tid] += red[tid + st]; __syncthreads(); }
  float inv = rsqrtf(red[0] * (1.0f / C_) + 1e-5f);
  for (int c = tid; c < C_; c += 256)
    out[(long long)row * C_ + c] = (x[c] - mean) * inv * g[c] + b[c];
}

__global__ __launch_bounds__(256)
void softmax_kernel(float* __restrict__ s) {   // rows of length NT_
  int row = blockIdx.x, tid = threadIdx.x;
  __shared__ float red[256];
  float* x = s + (long long)row * NT_;
  float m = -1e30f;
  for (int t = tid; t < NT_; t += 256) m = fmaxf(m, x[t]);
  red[tid] = m; __syncthreads();
  for (int st = 128; st > 0; st >>= 1) { if (tid < st) red[tid] = fmaxf(red[tid], red[tid + st]); __syncthreads(); }
  m = red[0]; __syncthreads();
  float sum = 0.0f;
  for (int t = tid; t < NT_; t += 256) { float e = __expf(x[t] - m); x[t] = e; sum += e; }
  red[tid] = sum; __syncthreads();
  for (int st = 128; st > 0; st >>= 1) { if (tid < st) red[tid] += red[tid + st]; __syncthreads(); }
  float inv = 1.0f / red[0];
  for (int t = tid; t < NT_; t += 256) x[t] *= inv;
}

// ------------------------- generic bf16 WMMA GEMM --------------------------
// C = act( alpha * A(f32,[M,K]) x B(bf16,[K,N] or [N,K]^T) + bias + residual )
// 256 threads = 8 wave32; block tile 16(M) x 128(N); K stepped by 32.
// LDS tiles are stored directly in WMMA *operand order* ([lane][16 halves]),
// so each lane fetches its v16bf with two ds_load_b128; staging writes are
// packed-u32 (v_cvt_pk_bf16_f32). N is required to be a multiple of 128 and
// K a multiple of 32 (true for every call here); M guard only when GUARD_M.

template <int TRANS_B, int GUARD_M>
__global__ __launch_bounds__(256)
void gemm_bf16(const float* __restrict__ A, long long sA1, long long sA2, int lda,
               const __bf16* __restrict__ B, long long sB1, long long sB2, int ldb,
               float* __restrict__ Cf, __bf16* __restrict__ Cb,
               long long sC1, long long sC2, int ldc,
               const float* __restrict__ bias,
               const float* __restrict__ res, long long sR1, long long sR2, int ldr,
               int M, int N, int K, int b2, float alpha, int act) {
  __shared__ __align__(32) uint32_t aOp32[32 * 8];        // 32 lanes x 16 bf16
  __shared__ __align__(32) uint32_t bOp32[8 * 32 * 8];    // 8 waves x 32 lanes x 16 bf16
  int z = blockIdx.z, z1 = z / b2, z2 = z % b2;
  A += z1 * sA1 + z2 * sA2;
  B += z1 * sB1 + z2 * sB2;
  long long coff = (long long)z1 * sC1 + z2 * sC2;
  if (res) res += z1 * sR1 + z2 * sR2;
  int m0 = blockIdx.y * 16;
  int n0 = blockIdx.x * 128;
  int tid = threadIdx.x, wave = tid >> 5, lane = tid & 31;
  v8f acc = {0.f, 0.f, 0.f, 0.f, 0.f, 0.f, 0.f, 0.f};
  // A staging: one packed u32 (2 halves along K) per thread, operand order.
  int am = tid >> 4;                    // row 0..15
  int ak = (tid & 15) * 2;              // even k 0..30
  int alane = am + ((ak & 16) ? 16 : 0);
  int ae = ak & 15;
  uint32_t* aDst = &aOp32[alane * 8 + (ae >> 1)];
  const float* aSrcBase = A + (long long)(m0 + am) * lda + ak;
  bool aValid = !GUARD_M || (m0 + am) < M;

  for (int k0 = 0; k0 < K; k0 += 32) {
    float lo = 0.0f, hi = 0.0f;
    if (aValid) { const float* ap = aSrcBase + k0; lo = ap[0]; hi = ap[1]; }
    *aDst = pack_bf16(lo, hi);
    if (TRANS_B) {
      #pragma unroll
      for (int i = 0; i < 8; ++i) {
        int j = tid + i * 256;          // 0..2047 packed u32 jobs
        int n = j >> 4;                 // 0..127
        int k = (j & 15) * 2;           // even
        uint32_t v = *(const uint32_t*)(B + (long long)(n0 + n) * ldb + k0 + k);
        int lb = (n & 15) + ((k & 16) ? 16 : 0);
        bOp32[((n >> 4) * 32 + lb) * 8 + ((k & 15) >> 1)] = v;
      }
    } else {
      __bf16* bH = (__bf16*)bOp32;
      #pragma unroll
      for (int i = 0; i < 8; ++i) {
        int j = tid + i * 256;
        int k = j >> 6;                 // 0..31
        int n = (j & 63) * 2;           // even
        union { uint32_t u; __bf16 b[2]; } cv;
        cv.u = *(const uint32_t*)(B + (long long)(k0 + k) * ldb + n0 + n);
        int hi16 = (k & 16) ? 16 : 0;
        int base = ((n >> 4) * 32 + (n & 15) + hi16) * 16 + (k & 15);
        bH[base]      = cv.b[0];
        bH[base + 16] = cv.b[1];        // lane+1, same e
      }
    }
    __syncthreads();
    const v16bf* av = (const v16bf*)((const __bf16*)aOp32 + lane * 16);
    const v16bf* bv = (const v16bf*)((const __bf16*)bOp32 + (wave * 32 + lane) * 16);
    acc = __builtin_amdgcn_wmma_f32_16x16x32_bf16(false, *av, false, *bv,
                                                  (short)0, acc, false, false);
    __syncthreads();
  }
  int nl = lane & 15, mh = (lane >> 4) << 3;
  int gn = n0 + wave * 16 + nl;
  #pragma unroll
  for (int v = 0; v < 8; ++v) {
    int gm = m0 + mh + v;
    if (!GUARD_M || gm < M) {
      float x = acc[v] * alpha;
      if (bias) x += bias[gn];
      if (res)  x += res[(long long)gm * ldr + gn];
      if (act == 1) x = 0.5f * x * (1.0f + erff(x * 0.70710678118654752f));
      long long o = coff + (long long)gm * ldc + gn;
      if (Cf) Cf[o] = x;
      if (Cb) Cb[o] = (__bf16)x;
    }
  }
}

// ------------------------------ host side ----------------------------------

static void launch_gemm(hipStream_t st, bool transB,
                        const float* A, long long sA1, long long sA2, int lda,
                        const __bf16* B, long long sB1, long long sB2, int ldb,
                        float* Cf, __bf16* Cb, long long sC1, long long sC2, int ldc,
                        const float* bias,
                        const float* res, long long sR1, long long sR2, int ldr,
                        int M, int N, int K, int b1, int b2, float alpha, int act) {
  dim3 grid(N / 128, (M + 15) / 16, b1 * b2);
  if (transB)
    gemm_bf16<1, 0><<<grid, 256, 0, st>>>(A, sA1, sA2, lda, B, sB1, sB2, ldb,
                                          Cf, Cb, sC1, sC2, ldc, bias,
                                          res, sR1, sR2, ldr, M, N, K, b2, alpha, act);
  else if (M % 16)
    gemm_bf16<0, 1><<<grid, 256, 0, st>>>(A, sA1, sA2, lda, B, sB1, sB2, ldb,
                                          Cf, Cb, sC1, sC2, ldc, bias,
                                          res, sR1, sR2, ldr, M, N, K, b2, alpha, act);
  else
    gemm_bf16<0, 0><<<grid, 256, 0, st>>>(A, sA1, sA2, lda, B, sB1, sB2, ldb,
                                          Cf, Cb, sC1, sC2, ldc, bias,
                                          res, sR1, sR2, ldr, M, N, K, b2, alpha, act);
}

static void conv(hipStream_t st, const float* in, __bf16* out, long long n) {
  f32_to_bf16_kernel<<<(unsigned)((n + 255) / 256), 256, 0, st>>>(in, out, n);
}

extern "C" void kernel_launch(void* const* d_in, const int* in_sizes, int n_in,
                              void* d_out, int out_size, void* d_ws, size_t ws_size,
                              hipStream_t stream) {
  (void)in_sizes; (void)n_in; (void)out_size; (void)ws_size;
  const float* feats   = (const float*)d_in[0];
  const int*   masks   = (const int*)  d_in[1];
  const float* localf  = (const float*)d_in[2];
  const float* globalf = (const float*)d_in[3];
  const float* boxes   = (const float*)d_in[4];
  /* d_in[5] = mask_num == 16 (hardcoded K_) */
  const float* pos_w = (const float*)d_in[6];
  const float* pos_b = (const float*)d_in[7];
  const float* lnl_g = (const float*)d_in[8];
  const float* lnl_b = (const float*)d_in[9];
  const float* lng_g = (const float*)d_in[10];
  const float* lng_b = (const float*)d_in[11];
  const float* fc1_w = (const float*)d_in[12];
  const float* fc1_b = (const float*)d_in[13];
  const float* fc2_w = (const float*)d_in[14];
  const float* fc2_b = (const float*)d_in[15];
  const float* wq_l = (const float*)d_in[16]; const float* bq_l = (const float*)d_in[17];
  const float* wk_l = (const float*)d_in[18]; /* bk cancels in softmax */
  const float* wv_l = (const float*)d_in[20]; const float* bv_l = (const float*)d_in[21];
  const float* wo_l = (const float*)d_in[22]; const float* bo_l = (const float*)d_in[23];
  const float* wq_g = (const float*)d_in[24]; const float* bq_g = (const float*)d_in[25];
  const float* wk_g = (const float*)d_in[26];
  const float* wv_g = (const float*)d_in[28]; const float* bv_g = (const float*)d_in[29];
  const float* wo_g = (const float*)d_in[30]; const float* bo_g = (const float*)d_in[31];
  float* outp = (float*)d_out;

  // workspace bump allocator (~415 MB total)
  char* p = (char*)d_ws;
  auto alloc = [&](size_t bytes) -> void* {
    void* q = (void*)p; p += (bytes + 255) & ~((size_t)255); return q;
  };
  float*  tokens = (float*) alloc((size_t)R_ * NT_ * C_ * 4);
  __bf16* kvlbf  = (__bf16*)alloc((size_t)R_ * NT_ * C_ * 2);
  __bf16* kvgbf  = (__bf16*)alloc((size_t)R_ * NT_ * C_ * 2);
  float*  uvals  = (float*) alloc((size_t)R_ * NT_ * 4);
  int*    labels = (int*)   alloc((size_t)R_ * NT_ * 4);
  float*  cent   = (float*) alloc((size_t)R_ * K_ * C_ * 4);
  float*  sums   = (float*) alloc((size_t)R_ * K_ * C_ * 4);
  float*  cnts   = (float*) alloc((size_t)R_ * K_ * 4);
  float*  ln_out = (float*) alloc((size_t)R_ * K_ * C_ * 4);
  float*  qproj  = (float*) alloc((size_t)R_ * K_ * C_ * 4);
  float*  Bmat   = (float*) alloc((size_t)R_ * H_ * K_ * C_ * 4);
  float*  scores = (float*) alloc((size_t)R_ * H_ * K_ * NT_ * 4);
  float*  AKV    = (float*) alloc((size_t)R_ * K_ * H_ * C_ * 4);
  __bf16* BigWbf = (__bf16*)alloc((size_t)H_ * C_ * C_ * 2);
  __bf16* wqbf   = (__bf16*)alloc((size_t)C_ * C_ * 2);
  __bf16* wkbf   = (__bf16*)alloc((size_t)C_ * C_ * 2);
  __bf16* wobf   = (__bf16*)alloc((size_t)C_ * C_ * 2);
  __bf16* fc1bf  = (__bf16*)alloc((size_t)C_ * HID_ * 2);
  __bf16* fc2bf  = (__bf16*)alloc((size_t)HID_ * HID_ * 2);
  float*  fused  = (float*) alloc((size_t)R_ * K_ * C_ * 4);
  float*  fused2 = (float*) alloc((size_t)R_ * K_ * C_ * 4);
  float*  h1     = (float*) alloc((size_t)R_ * K_ * HID_ * 4);
  float*  bvo    = (float*) alloc((size_t)C_ * 4);

  const long long NRC = (long long)R_ * NT_ * C_;

  // bf16 copies of the streamed KV tensors and FFN weights (halve HBM traffic)
  conv(stream, localf, kvlbf, NRC);
  conv(stream, globalf, kvgbf, NRC);
  conv(stream, fc1_w, fc1bf, (long long)C_ * HID_);
  conv(stream, fc2_w, fc2bf, (long long)HID_ * HID_);

  // tokens = feats + pos-linear
  pos_tokens_kernel<<<(unsigned)((NRC + 255) / 256), 256, 0, stream>>>(
      feats, boxes, pos_w, pos_b, tokens, NRC);

  // JAX-threefry uniforms, top-16 init, 3 k-means iterations
  uniform_kernel<<<R_, 256, 0, stream>>>(masks, uvals);
  topk_init_kernel<<<R_, 256, 0, stream>>>(uvals, tokens, cent);
  for (int it = 0; it < 3; ++it) {
    kmeans_assign_kernel<<<dim3(R_, NT_ / 256), 256, 0, stream>>>(tokens, cent, labels);
    (void)hipMemsetAsync(sums, 0, (size_t)R_ * K_ * C_ * 4, stream);
    (void)hipMemsetAsync(cnts, 0, (size_t)R_ * K_ * 4, stream);
    kmeans_update_kernel<<<dim3(R_, 8, 4), 128, 0, stream>>>(tokens, labels, masks, sums, cnts);
    cent_update_kernel<<<(R_ * K_ * C_) / 256, 256, 0, stream>>>(sums, cnts, cent, R_ * K_ * C_);
  }

  // fused = qin + MHA(LN(qin), kv)  using low-rank refactoring:
  //   Bmat_h = (LN q @ Wq)_h @ Wk_h^T / sqrt(dh);  S = Bmat @ KV^T (bk drops)
  //   out    = (A @ KV) @ (Wv_h Wo_h) + bv@Wo + bo  (softmax rows sum to 1)
  auto run_mha = [&](const float* qin, const float* lng, const float* lnb,
                     const float* wq, const float* bq, const float* wk,
                     const float* wv, const float* bv, const float* wo,
                     const float* bo, const __bf16* kvbf, float* outbuf) {
    conv(stream, wq, wqbf, (long long)C_ * C_);
    conv(stream, wk, wkbf, (long long)C_ * C_);
    conv(stream, wo, wobf, (long long)C_ * C_);
    layernorm_kernel<<<R_ * K_, 256, 0, stream>>>(qin, lng, lnb, ln_out);
    // qproj = LN @ Wq + bq   [256,1024]
    launch_gemm(stream, false, ln_out, 0, 0, C_, wqbf, 0, 0, C_,
                qproj, nullptr, 0, 0, C_, bq, nullptr, 0, 0, 0,
                R_ * K_, C_, C_, 1, 1, 1.0f, 0);
    // Bmat[r,h] = qproj_h @ Wk_h^T * 1/sqrt(dh)   [16,1024] per (r,h)
    launch_gemm(stream, true, qproj, (long long)K_ * C_, DH_, C_,
                wkbf, 0, DH_, C_,
                Bmat, nullptr, (long long)H_ * K_ * C_, (long long)K_ * C_, C_,
                nullptr, nullptr, 0, 0, 0,
                K_, C_, DH_, R_, H_, 0.08838834764831845f, 0);
    // scores[r,h] = Bmat @ KV^T   [16,2304]
    launch_gemm(stream, true, Bmat, (long long)H_ * K_ * C_, (long long)K_ * C_, C_,
                kvbf, (long long)NT_ * C_, 0, C_,
                scores, nullptr, (long long)H_ * K_ * NT_, (long long)K_ * NT_, NT_,
                nullptr, nullptr, 0, 0, 0,
                K_, NT_, C_, R_, H_, 1.0f, 0);
    softmax_kernel<<<R_ * H_ * K_, 256, 0, stream>>>(scores);
    // AKV[r,q,(h,c)] = A @ KV   (stored [R,16,H*1024])
    launch_gemm(stream, false, scores, (long long)H_ * K_ * NT_, (long long)K_ * NT_, NT_,
                kvbf, (long long)NT_ * C_, 0, C_,
                AKV, nullptr, (long long)K_ * H_ * C_, C_, H_ * C_,
                nullptr, nullptr, 0, 0, 0,
                K_, C_, NT_, R_, H_, 1.0f, 0);
    // BigW_h = Wv_h @ Wo_h  -> bf16 [8192,1024]
    launch_gemm(stream, false, wv, DH_, 0, C_,
                wobf, (long long)DH_ * C_, 0, C_,
                nullptr, BigWbf, (long long)C_ * C_, 0, C_,
                nullptr, nullptr, 0, 0, 0,
                C_, C_, DH_, H_, 1, 1.0f, 0);
    // bvo = bv @ Wo + bo
    launch_gemm(stream, false, bv, 0, 0, C_, wobf, 0, 0, C_,
                bvo, nullptr, 0, 0, C_, bo, nullptr, 0, 0, 0,
                1, C_, C_, 1, 1, 1.0f, 0);
    // out = qin + AKV @ BigW + bvo
    launch_gemm(stream, false, AKV, 0, 0, H_ * C_, BigWbf, 0, 0, C_,
                outbuf, nullptr, 0, 0, C_, bvo, qin, 0, 0, C_,
                R_ * K_, C_, H_ * C_, 1, 1, 1.0f, 0);
  };

  run_mha(cent,  lnl_g, lnl_b, wq_l, bq_l, wk_l, wv_l, bv_l, wo_l, bo_l, kvlbf, fused);
  run_mha(fused, lng_g, lng_b, wq_g, bq_g, wk_g, wv_g, bv_g, wo_g, bo_g, kvgbf, fused2);

  // FFN: GELU(x @ fc1 + b1) @ fc2 + b2  -> d_out [256, 4096]
  launch_gemm(stream, false, fused2, 0, 0, C_, fc1bf, 0, 0, HID_,
              h1, nullptr, 0, 0, HID_, fc1_b, nullptr, 0, 0, 0,
              R_ * K_, HID_, C_, 1, 1, 1.0f, 1);
  launch_gemm(stream, false, h1, 0, 0, HID_, fc2bf, 0, 0, HID_,
              outp, nullptr, 0, 0, HID_, fc2_b, nullptr, 0, 0, 0,
              R_ * K_, HID_, HID_, 1, 1, 1.0f, 0);
}
